// SGCN_70454643524122
// MI455X (gfx1250) — matrix operand order repaired and checked by reading, not compiled, not run on hardware
//
#include <hip/hip_runtime.h>
#include <hip/hip_bf16.h>

typedef __attribute__((ext_vector_type(16))) __bf16 v16bf;
typedef __attribute__((ext_vector_type(8)))  float  v8f;

#define GDIM 128
#define HDIM 128

__device__ __forceinline__ __bf16 tobf(float f) { return (__bf16)f; }

// ---------------------------------------------------------------------------
// Kernel 0: init per-node softmax state
// ---------------------------------------------------------------------------
__global__ __launch_bounds__(256)
void init_nodes(float* __restrict__ nmax, float* __restrict__ nsum, int n) {
    int i = blockIdx.x * 256 + threadIdx.x;
    if (i < n) { nmax[i] = -3.402823466e38f; nsum[i] = 0.0f; }
}

// ---------------------------------------------------------------------------
// Kernel 1: WMMA GEMM  dst = x @ W + b   (one of q/k/v/skip per blockIdx.y)
// Block: 256 threads = 8 waves; each wave does a 16-row x 128-col strip.
// Weights staged in LDS as bf16, PRE-SWIZZLED into the WMMA B-fragment
// per-lane layout so each fragment is one contiguous 32-byte LDS read.
//
// Fragment layout (V_WMMA_F32_16X16X32_BF16, B = 32x16):
//   lane 0-15  : col = lane,     K = kt*32 + {0..7, 16..23}
//   lane 16-31 : col = lane-16,  K = kt*32 + {8..15, 24..31}   (2 K per VGPR)
// ---------------------------------------------------------------------------
__global__ __launch_bounds__(256)
void qkvs_gemm(const float* __restrict__ x,
               const float* __restrict__ Wq, const float* __restrict__ bq,
               const float* __restrict__ Wk, const float* __restrict__ bk,
               const float* __restrict__ Wv, const float* __restrict__ bv,
               const float* __restrict__ Ws, const float* __restrict__ bs,
               float* __restrict__ q, float* __restrict__ k,
               float* __restrict__ v, float* __restrict__ outskip,
               int nnodes)
{
    __shared__ __bf16 Wswz[GDIM * HDIM];           // 32 KB swizzled bf16

    const int tid = threadIdx.x;
    const int mat = blockIdx.y;                    // 0=q 1=k 2=v 3=skip

    const float* W;  const float* bias;  float* dst;
    switch (mat) {
        case 0:  W = Wq; bias = bq; dst = q;       break;
        case 1:  W = Wk; bias = bk; dst = k;       break;
        case 2:  W = Wv; bias = bv; dst = v;       break;
        default: W = Ws; bias = bs; dst = outskip; break;
    }

    // Stage + swizzle W (fp32 -> bf16). Global reads coalesced; LDS writes
    // scattered (scatter-tolerant side).
    for (int i = tid; i < GDIM * HDIM; i += 256) {
        const int kk = i >> 7;                 // K row of W: 0..127
        const int c  = i & 127;                // output col: 0..127
        const int kt = kk >> 5;                // K-tile 0..3
        const int kr = kk & 31;                // K within tile
        const int frag = ((c >> 4) << 2) | kt; // nt*4 + kt
        const int ln   = (((kr >> 3) & 1) << 4) | (c & 15);
        Wswz[(((frag << 5) | ln) << 4) | (((kr >> 4) << 3) | (kr & 7))] = tobf(W[i]);
    }
    __syncthreads();

    const int wave = tid >> 5;                     // 0..7
    const int lane = tid & 31;
    const int rowBase = blockIdx.x * 128 + wave * 16;
    if (rowBase >= nnodes) return;                 // wave-uniform early out

    // --- A fragments: 16x32 bf16, 4 K-tiles (K = 0..127) ---
    // lanes 0-15 : M=lane,    K = kt*32 + {0..7, 16..23}
    // lanes 16-31: M=lane-16, K = kt*32 + {8..15, 24..31}
    int arow = rowBase + (lane & 15);
    if (arow >= nnodes) arow = nnodes - 1;         // clamp (stores are guarded)
    const int    kh   = (lane >> 4) * 8;
    const float* xrow = x + (size_t)arow * GDIM;

    v16bf afrag[4];
#pragma unroll
    for (int kt = 0; kt < 4; ++kt) {
        const int k0 = kt * 32 + kh;
        const float4 a0 = *(const float4*)(xrow + k0);
        const float4 a1 = *(const float4*)(xrow + k0 + 4);
        const float4 a2 = *(const float4*)(xrow + k0 + 16);
        const float4 a3 = *(const float4*)(xrow + k0 + 20);
        afrag[kt][0]  = tobf(a0.x); afrag[kt][1]  = tobf(a0.y);
        afrag[kt][2]  = tobf(a0.z); afrag[kt][3]  = tobf(a0.w);
        afrag[kt][4]  = tobf(a1.x); afrag[kt][5]  = tobf(a1.y);
        afrag[kt][6]  = tobf(a1.z); afrag[kt][7]  = tobf(a1.w);
        afrag[kt][8]  = tobf(a2.x); afrag[kt][9]  = tobf(a2.y);
        afrag[kt][10] = tobf(a2.z); afrag[kt][11] = tobf(a2.w);
        afrag[kt][12] = tobf(a3.x); afrag[kt][13] = tobf(a3.y);
        afrag[kt][14] = tobf(a3.z); afrag[kt][15] = tobf(a3.w);
    }

    const v16bf* Wfrag = (const v16bf*)Wswz;       // 32 bytes per lane-frag
    const int col  = lane & 15;
    const int rtop = rowBase + ((lane >> 4) << 3); // C layout: M=r / r+8
    const bool fullStrip = (rowBase + 16) <= nnodes; // wave-uniform

#pragma unroll
    for (int nt = 0; nt < 8; ++nt) {
        const int cbase = nt * 16;
        v8f acc = {};
#pragma unroll
        for (int kt = 0; kt < 4; ++kt) {
            const v16bf bfr = Wfrag[(((nt << 2) | kt) << 5) | lane];
            acc = __builtin_amdgcn_wmma_f32_16x16x32_bf16(
                false, afrag[kt], false, bfr,
                (short)0, acc, false, false);
        }
        const float bcol = bias[cbase + col];
        float* po = dst + (size_t)rtop * HDIM + cbase + col;
        if (fullStrip) {
#pragma unroll
            for (int r = 0; r < 8; ++r)
                po[r * HDIM] = acc[r] + bcol;      // one base + imm offsets
        } else {
#pragma unroll
            for (int r = 0; r < 8; ++r)
                if (rtop + r < nnodes) po[r * HDIM] = acc[r] + bcol;
        }
    }
}

// ---------------------------------------------------------------------------
// Kernel 2: per-edge logits = (q[dst] . k[src]) / sqrt(H); atomic max per dst
// ---------------------------------------------------------------------------
__global__ __launch_bounds__(256)
void edge_logits(const float* __restrict__ q, const float* __restrict__ k,
                 const int* __restrict__ src, const int* __restrict__ dst,
                 float* __restrict__ logits, float* __restrict__ nmax, int E)
{
    int e = blockIdx.x * 256 + threadIdx.x;
    if (e >= E) return;
    const int s = src[e], d = dst[e];
    const float4* qd = (const float4*)(q + (size_t)d * HDIM);
    const float4* ks = (const float4*)(k + (size_t)s * HDIM);
    float acc = 0.0f;
#pragma unroll
    for (int j = 0; j < HDIM / 4; ++j) {
        float4 a = qd[j], b = ks[j];
        acc += a.x * b.x + a.y * b.y + a.z * b.z + a.w * b.w;
    }
    acc *= 0.08838834764831845f;                   // 1/sqrt(128)
    logits[e] = acc;
    __hip_atomic_fetch_max(nmax + d, acc, __ATOMIC_RELAXED,
                           __HIP_MEMORY_SCOPE_AGENT);
}

// ---------------------------------------------------------------------------
// Kernel 3: p = exp(logit - max[dst]) (in place); atomic sum per dst
// ---------------------------------------------------------------------------
__global__ __launch_bounds__(256)
void edge_exp(float* __restrict__ logits, const int* __restrict__ dst,
              const float* __restrict__ nmax, float* __restrict__ nsum, int E)
{
    int e = blockIdx.x * 256 + threadIdx.x;
    if (e >= E) return;
    const int d = dst[e];
    const float p = __expf(logits[e] - nmax[d]);
    logits[e] = p;                                 // overwrite logits with p
    atomicAdd(nsum + d, p);
}

// ---------------------------------------------------------------------------
// Kernel 4: one wave32 per edge: out[dst,:] += (p/denom[dst]) * v[src,:]
// 4 floats per lane covers H=128.
// ---------------------------------------------------------------------------
__global__ __launch_bounds__(256)
void edge_scatter(const float* __restrict__ p, const float* __restrict__ nsum,
                  const float* __restrict__ v,
                  const int* __restrict__ src, const int* __restrict__ dst,
                  float* __restrict__ out, int E)
{
    const int gid  = blockIdx.x * 256 + threadIdx.x;
    const int e    = gid >> 5;
    const int lane = threadIdx.x & 31;
    if (e >= E) return;
    const int s = src[e], d = dst[e];
    const float alpha = p[e] / nsum[d];
    const float4 vv = ((const float4*)(v + (size_t)s * HDIM))[lane];
    float* o = out + (size_t)d * HDIM + lane * 4;
    atomicAdd(o + 0, alpha * vv.x);
    atomicAdd(o + 1, alpha * vv.y);
    atomicAdd(o + 2, alpha * vv.z);
    atomicAdd(o + 3, alpha * vv.w);
}

// ---------------------------------------------------------------------------
extern "C" void kernel_launch(void* const* d_in, const int* in_sizes, int n_in,
                              void* d_out, int out_size, void* d_ws, size_t ws_size,
                              hipStream_t stream)
{
    (void)n_in; (void)out_size; (void)ws_size;
    const float* x    = (const float*)d_in[0];
    const int*   eidx = (const int*)  d_in[1];   // [2, E]
    // d_in[2] edge_norm, d_in[3] edge_type: unused by reference forward
    const float* Wq = (const float*)d_in[4];  const float* bq = (const float*)d_in[5];
    const float* Wk = (const float*)d_in[6];  const float* bk = (const float*)d_in[7];
    const float* Wv = (const float*)d_in[8];  const float* bv = (const float*)d_in[9];
    const float* Ws = (const float*)d_in[10]; const float* bs = (const float*)d_in[11];

    const int N = in_sizes[0] / GDIM;
    const int E = in_sizes[1] / 2;
    const int* src = eidx;
    const int* dst = eidx + E;

    float* out = (float*)d_out;

    // workspace layout: q | k | v | logits/p | node_max | node_sum
    float* q      = (float*)d_ws;
    float* k      = q + (size_t)N * HDIM;
    float* v      = k + (size_t)N * HDIM;
    float* logits = v + (size_t)N * HDIM;
    float* nmax   = logits + E;
    float* nsum   = nmax + N;

    init_nodes<<<(N + 255) / 256, 256, 0, stream>>>(nmax, nsum, N);

    dim3 g1((N + 127) / 128, 4);
    qkvs_gemm<<<g1, 256, 0, stream>>>(x, Wq, bq, Wk, bk, Wv, bv, Ws, bs,
                                      q, k, v, out, N);

    edge_logits<<<(E + 255) / 256, 256, 0, stream>>>(q, k, src, dst,
                                                     logits, nmax, E);

    edge_exp<<<(E + 255) / 256, 256, 0, stream>>>(logits, dst, nmax, nsum, E);

    const long long threads = (long long)E * 32;
    edge_scatter<<<(unsigned)((threads + 255) / 256), 256, 0, stream>>>(
        logits, nsum, v, src, dst, out, E);
}
